// DualBranchExtractor_51049981280566
// MI455X (gfx1250) — matrix operand order repaired and checked by reading, not compiled, not run on hardware
//
#include <hip/hip_runtime.h>
#include <math.h>

// ---------------- problem constants (match reference) ----------------
#define BB   32768
#define DD   1024
#define HH   16
#define HD   64
#define LN_EPS 1e-5f

// ---------------- WMMA vector types (gfx1250, wave32) ----------------
typedef __attribute__((ext_vector_type(16))) __bf16 v16bf;
typedef __attribute__((ext_vector_type(8)))  __bf16 v8bf;
typedef __attribute__((ext_vector_type(8)))  float  v8f;

// =====================================================================
// Tiled bf16 WMMA GEMM:  C[M,N] = A[M,K] @ W[N,K]^T + bias[N]
//   A, W row-major bf16 (K contiguous)  ->  contiguous k = 16*lhi + s, same for A and B
//   block = 256 threads = 8 waves; wave tile = 32(M) x 128(N)
//     -> 2 A-frags x 8 B-frags = 16 v_wmma per K-step, each B frag
//        reused twice (16 WMMA : 20 b128 loads instead of 8 : 18)
//   block tile = 256 x 128 ; K stepped by 32 (v_wmma_f32_16x16x32_bf16)
// EPI: 0 = store bf16, 1 = store f32, 2 = exact GELU then store f32
// =====================================================================
template <int EPI>
__global__ __launch_bounds__(256)
void gemm_bf16_wmma(const __bf16* __restrict__ A,
                    const __bf16* __restrict__ W,
                    const float*  __restrict__ bias,
                    void* __restrict__ Cout,
                    int M, int N, int K)
{
    const int lane = threadIdx.x & 31;
    const int wave = threadIdx.x >> 5;
    const int l15  = lane & 15;
    const int lhi  = lane >> 4;          // 0: lanes 0-15, 1: lanes 16-31

    const int rowT = blockIdx.y * 256 + wave * 32;  // wave's 32 M rows
    const int colT = blockIdx.x * 128;              // block's 128 N cols

    v8f acc[2][8];
    const v8f vzero = {0.f,0.f,0.f,0.f,0.f,0.f,0.f,0.f};
#pragma unroll
    for (int i = 0; i < 2; ++i)
#pragma unroll
        for (int j = 0; j < 8; ++j) acc[i][j] = vzero;

    // A fragment base: lane (0-15) -> row l15, K chunk [0..15]
    //                  lane (16-31)-> row l15, K chunk [16..31]
    const __bf16* aptr0 = A + (size_t)(rowT + l15)      * (size_t)K + (size_t)(lhi * 16);
    const __bf16* aptr1 = A + (size_t)(rowT + 16 + l15) * (size_t)K + (size_t)(lhi * 16);
    // B fragment base: lane (0-15) -> W row (col) l15, K [0..15]
    //                  lane (16-31)-> W row (col) l15, K [16..31]
    const __bf16* bbase = W + (size_t)(colT + l15) * (size_t)K + (size_t)(lhi * 16);

    for (int k0 = 0; k0 < K; k0 += 32) {
        union { v16bf v; v8bf h[2]; } a0, a1;
        a0.h[0] = *(const v8bf*)(aptr0 + k0);
        a0.h[1] = *(const v8bf*)(aptr0 + k0 + 8);
        a1.h[0] = *(const v8bf*)(aptr1 + k0);
        a1.h[1] = *(const v8bf*)(aptr1 + k0 + 8);
#pragma unroll
        for (int j = 0; j < 8; ++j) {
            v16bf bf = *(const v16bf*)(bbase + (size_t)j * 16u * (size_t)K + k0); // 32B load
            acc[0][j] = __builtin_amdgcn_wmma_f32_16x16x32_bf16(
                            false, a0.v, false, bf, (short)0, acc[0][j], false, false);
            acc[1][j] = __builtin_amdgcn_wmma_f32_16x16x32_bf16(
                            false, a1.v, false, bf, (short)0, acc[1][j], false, false);
        }
    }

    // Epilogue. C/D layout: VGPR r -> row (lhi*8 + r); lane l15 -> col.
#pragma unroll
    for (int j = 0; j < 8; ++j) {
        const int col = colT + j * 16 + l15;
        const float bv = bias[col];
#pragma unroll
        for (int i = 0; i < 2; ++i) {
#pragma unroll
            for (int r = 0; r < 8; ++r) {
                const int row = rowT + i * 16 + lhi * 8 + r;
                float v = acc[i][j][r] + bv;
                if (EPI == 0) {
                    ((__bf16*)Cout)[(size_t)row * (size_t)N + col] = (__bf16)v;
                } else if (EPI == 1) {
                    ((float*)Cout)[(size_t)row * (size_t)N + col] = v;
                } else {
                    float g = 0.5f * v * (1.0f + erff(v * 0.70710678118654752f));
                    ((float*)Cout)[(size_t)row * (size_t)N + col] = g;
                }
            }
        }
    }
}

// ------------------- f32 -> bf16 conversion -------------------
__global__ void cvt_f32_bf16(const float* __restrict__ in,
                             __bf16* __restrict__ out, size_t n)
{
    size_t i = (size_t)blockIdx.x * blockDim.x + threadIdx.x;
    if (i < n) out[i] = (__bf16)in[i];
}

// qact = bf16(Z_T)  or  bf16(Z_T * (1 + S[row]))
__global__ void make_qact(const float* __restrict__ zt,
                          const float* __restrict__ S,
                          __bf16* __restrict__ out, int useS)
{
    size_t i = (size_t)blockIdx.x * blockDim.x + threadIdx.x;
    if (i < (size_t)BB * DD) {
        float v = zt[i];
        if (useS) v *= (1.0f + S[i / DD]);
        out[i] = (__bf16)v;
    }
}

// ---------------- 2-key attention (seq len 2, per (b,h)) ----------------
// K,V stored as [2B, D]: rows 0..B-1 from Z_V, rows B..2B-1 from Z_A.
__global__ void attn2(const __bf16* __restrict__ Q,
                      const __bf16* __restrict__ Kc,
                      const __bf16* __restrict__ Vc,
                      __bf16* __restrict__ AO)
{
    size_t t = (size_t)blockIdx.x * blockDim.x + threadIdx.x;
    if (t >= (size_t)BB * HH) return;
    const int b = (int)(t / HH);
    const int h = (int)(t % HH);
    const size_t base  = (size_t)b * DD + (size_t)h * HD;
    const size_t base1 = base + (size_t)BB * DD;

    float s0 = 0.f, s1 = 0.f;
#pragma unroll 4
    for (int d = 0; d < HD; ++d) {
        float q = (float)Q[base + d];
        s0 += q * (float)Kc[base + d];
        s1 += q * (float)Kc[base1 + d];
    }
    s0 *= 0.125f; s1 *= 0.125f;         // 1/sqrt(64)
    float m  = fmaxf(s0, s1);
    float e0 = __expf(s0 - m), e1 = __expf(s1 - m);
    float inv = 1.0f / (e0 + e1);
    float a0 = e0 * inv, a1 = e1 * inv;
#pragma unroll 4
    for (int d = 0; d < HD; ++d) {
        float o = a0 * (float)Vc[base + d] + a1 * (float)Vc[base1 + d];
        AO[base + d] = (__bf16)o;
    }
}

// ---------------- LayerNorm over D, write bf16 ----------------
__global__ __launch_bounds__(256)
void layernorm_bf16(const float* __restrict__ X,
                    const float* __restrict__ gamma,
                    const float* __restrict__ beta,
                    __bf16* __restrict__ Y)
{
    const int row = blockIdx.x;
    const float* x = X + (size_t)row * DD;
    __shared__ float s1[256], s2[256];
    float ls = 0.f, lq = 0.f;
    for (int c = threadIdx.x; c < DD; c += 256) {
        float v = x[c]; ls += v; lq += v * v;
    }
    s1[threadIdx.x] = ls; s2[threadIdx.x] = lq;
    __syncthreads();
    for (int off = 128; off > 0; off >>= 1) {
        if ((int)threadIdx.x < off) {
            s1[threadIdx.x] += s1[threadIdx.x + off];
            s2[threadIdx.x] += s2[threadIdx.x + off];
        }
        __syncthreads();
    }
    const float mean = s1[0] * (1.0f / DD);
    const float var  = s2[0] * (1.0f / DD) - mean * mean;
    const float rstd = rsqrtf(var + LN_EPS);
    __bf16* y = Y + (size_t)row * DD;
    for (int c = threadIdx.x; c < DD; c += 256)
        y[c] = (__bf16)((x[c] - mean) * rstd * gamma[c] + beta[c]);
}

// =====================================================================
extern "C" void kernel_launch(void* const* d_in, const int* in_sizes, int n_in,
                              void* d_out, int out_size, void* d_ws, size_t ws_size,
                              hipStream_t stream)
{
    (void)in_sizes; (void)n_in; (void)out_size; (void)ws_size;
    const float* Z_T   = (const float*)d_in[0];
    const float* Z_V   = (const float*)d_in[1];
    const float* Z_A   = (const float*)d_in[2];
    const float* S     = (const float*)d_in[3];
    const float* Wqkv[2] = { (const float*)d_in[4],  (const float*)d_in[8]  };
    const float* bqkv[2] = { (const float*)d_in[5],  (const float*)d_in[9]  };
    const float* Wo[2]   = { (const float*)d_in[6],  (const float*)d_in[10] };
    const float* bo[2]   = { (const float*)d_in[7],  (const float*)d_in[11] };
    const float* Wp[2]   = { (const float*)d_in[12], (const float*)d_in[14] };
    const float* bp[2]   = { (const float*)d_in[13], (const float*)d_in[15] };
    const float* gamma = (const float*)d_in[16];
    const float* beta  = (const float*)d_in[17];
    float* out = (float*)d_out;

    const size_t bd = (size_t)BB * DD;
    const size_t dd = (size_t)DD * DD;

    // ---- workspace carve-out (256B aligned) ----
    size_t off = 0;
    auto carve = [&](size_t bytes) -> void* {
        void* p = (char*)d_ws + off;
        off += (bytes + 255) & ~(size_t)255;
        return p;
    };
    __bf16* ZVA  = (__bf16*)carve(2 * bd * 2);   // [Z_V; Z_A] bf16
    __bf16* qact = (__bf16*)carve(bd * 2);
    __bf16* Qb   = (__bf16*)carve(bd * 2);
    __bf16* Kb   = (__bf16*)carve(2 * bd * 2);
    __bf16* Vb   = (__bf16*)carve(2 * bd * 2);
    __bf16* AO   = (__bf16*)carve(bd * 2);
    float*  proj = (float*) carve(bd * 4);
    __bf16* xln  = (__bf16*)carve(bd * 2);
    __bf16* Wqkv_b[2], *Wo_b[2], *Wp_b[2];
    for (int br = 0; br < 2; ++br) {
        Wqkv_b[br] = (__bf16*)carve(3 * dd * 2);
        Wo_b[br]   = (__bf16*)carve(dd * 2);
        Wp_b[br]   = (__bf16*)carve(dd * 2);
    }

    const int T = 256;
    auto blocks = [&](size_t n) { return dim3((unsigned)((n + T - 1) / T)); };

    // ---- one-time conversions ----
    cvt_f32_bf16<<<blocks(bd), T, 0, stream>>>(Z_V, ZVA,      bd);
    cvt_f32_bf16<<<blocks(bd), T, 0, stream>>>(Z_A, ZVA + bd, bd);
    for (int br = 0; br < 2; ++br) {
        cvt_f32_bf16<<<blocks(3 * dd), T, 0, stream>>>(Wqkv[br], Wqkv_b[br], 3 * dd);
        cvt_f32_bf16<<<blocks(dd),     T, 0, stream>>>(Wo[br],   Wo_b[br],   dd);
        cvt_f32_bf16<<<blocks(dd),     T, 0, stream>>>(Wp[br],   Wp_b[br],   dd);
    }

    const dim3 gB (DD / 128, BB / 256);        // M = B     GEMM grid
    const dim3 g2B(DD / 128, 2 * BB / 256);    // M = 2B    GEMM grid

    for (int br = 0; br < 2; ++br) {
        // q activation (conf branch scales by 1+S)
        make_qact<<<blocks(bd), T, 0, stream>>>(Z_T, S, qact, br);

        // Q = qact @ Wq^T + bq ; K/V over stacked [Z_V;Z_A]
        gemm_bf16_wmma<0><<<gB,  T, 0, stream>>>(qact, Wqkv_b[br],          bqkv[br],          Qb, BB,     DD, DD);
        gemm_bf16_wmma<0><<<g2B, T, 0, stream>>>(ZVA,  Wqkv_b[br] + dd,     bqkv[br] + DD,     Kb, 2 * BB, DD, DD);
        gemm_bf16_wmma<0><<<g2B, T, 0, stream>>>(ZVA,  Wqkv_b[br] + 2 * dd, bqkv[br] + 2 * DD, Vb, 2 * BB, DD, DD);

        // softmax attention over the 2 keys
        attn2<<<blocks((size_t)BB * HH), T, 0, stream>>>(Qb, Kb, Vb, AO);

        // output projection (f32), LayerNorm, MLP + exact GELU -> d_out
        gemm_bf16_wmma<1><<<gB, T, 0, stream>>>(AO, Wo_b[br], bo[br], proj, BB, DD, DD);
        layernorm_bf16<<<BB, T, 0, stream>>>(proj, gamma, beta, xln);
        gemm_bf16_wmma<2><<<gB, T, 0, stream>>>(xln, Wp_b[br], bp[br],
                                                out + (size_t)br * bd, BB, DD, DD);
    }
}
